// RelationTransformerDecoderLayer_74388833566856
// MI455X (gfx1250) — compile-verified, matrix operand order checked by reading, not run on hardware
//
#include <hip/hip_runtime.h>
#include <hip/hip_bf16.h>

typedef __attribute__((ext_vector_type(16))) _Float16 v16h;
typedef __attribute__((ext_vector_type(8)))  _Float16 v8h;
typedef __attribute__((ext_vector_type(8)))  float    v8f;
typedef __attribute__((ext_vector_type(4)))  float    v4f;

#define Cdim 256
#define NHh  8
#define HDd  32
#define NLl  4
#define NPp  4
#define DFFd 1024
#define Bb   8
#define NQq  1000
#define NVv  13294
#define NDNn 100
#define GEMM_WAVES 8

// Load one 16-lane-striped A/B fragment (f16) from a row-major f32 row
// pointer that is already offset by the lane's half-wave K base (kb).
// Per lane: f[0..7] = p[0..7], f[8..15] = p[16..23].  4x global_load_b128.
__device__ __forceinline__ v16h frag_f32(const float* __restrict__ p) {
  const v4f a0 = *(const v4f*)(p);
  const v4f a1 = *(const v4f*)(p + 4);
  const v4f b0 = *(const v4f*)(p + 16);
  const v4f b1 = *(const v4f*)(p + 20);
  v16h f;
#pragma unroll
  for (int i = 0; i < 4; ++i) {
    f[i]      = (_Float16)a0[i];
    f[4 + i]  = (_Float16)a1[i];
    f[8 + i]  = (_Float16)b0[i];
    f[12 + i] = (_Float16)b1[i];
  }
  return f;
}

// ---------------------------------------------------------------------------
// WMMA GEMM: Cmat[M,N] = act(A[M,K] * W[N,K]^T + bias[N])
// One wave computes a 32x32 output block: 2 A-frags x 2 B-frags -> 4 WMMAs
// per 32-deep K-step (halves fetched bytes per matrix op vs 16x32).
// Row bounds handled by pointer clamping (garbage rows never stored).
// ---------------------------------------------------------------------------
__global__ void __launch_bounds__(256) gemm_wmma_kernel(
    const float* __restrict__ A, const float* __restrict__ W,
    const float* __restrict__ bias, float* __restrict__ Cmat,
    int M, int N, int K, int act)
{
  const int lane = threadIdx.x & 31;
  const int wave = threadIdx.x >> 5;
  const int tilesN = N >> 5;                   // 32-wide tiles
  const long tilesM = (long)(M + 31) >> 5;     // 32-high tiles
  const long tile = (long)blockIdx.x * GEMM_WAVES + wave;
  if (tile >= tilesM * (long)tilesN) return;
  const int tm = (int)(tile / tilesN);
  const int tn = (int)(tile % tilesN);
  const int row0 = tm << 5;
  const int col0 = tn << 5;
  const int r  = lane & 15;
  const int kb = (lane >> 4) << 3;

  int rA0 = row0 + r;       if (rA0 >= M) rA0 = M - 1;   // clamp: no predication
  int rA1 = row0 + 16 + r;  if (rA1 >= M) rA1 = M - 1;
  const float* ap0 = A + (long)rA0 * K + kb;
  const float* ap1 = A + (long)rA1 * K + kb;
  const float* wp0 = W + (long)(col0 + r) * K + kb;
  const float* wp1 = wp0 + (long)16 * K;

  v8f acc00 = {}, acc01 = {}, acc10 = {}, acc11 = {};
  for (int k0 = 0; k0 < K; k0 += 32) {
    if (k0 + 32 < K) {
      __builtin_prefetch(ap0 + k0 + 32, 0, 1);  // global_prefetch_b8
      __builtin_prefetch(ap1 + k0 + 32, 0, 1);
      __builtin_prefetch(wp0 + k0 + 32, 0, 1);
      __builtin_prefetch(wp1 + k0 + 32, 0, 1);
    }
    const v16h a0 = frag_f32(ap0 + k0);
    const v16h a1 = frag_f32(ap1 + k0);
    const v16h b0 = frag_f32(wp0 + k0);
    const v16h b1 = frag_f32(wp1 + k0);
    acc00 = __builtin_amdgcn_wmma_f32_16x16x32_f16(false, a0, false, b0,
                                                   (short)0, acc00, false, false);
    acc01 = __builtin_amdgcn_wmma_f32_16x16x32_f16(false, a0, false, b1,
                                                   (short)0, acc01, false, false);
    acc10 = __builtin_amdgcn_wmma_f32_16x16x32_f16(false, a1, false, b0,
                                                   (short)0, acc10, false, false);
    acc11 = __builtin_amdgcn_wmma_f32_16x16x32_f16(false, a1, false, b1,
                                                   (short)0, acc11, false, false);
  }

  const int rbase = row0 + ((lane >> 4) << 3);
  const int c0 = col0 + r;
  const int c1 = c0 + 16;
  const float bv0 = bias ? bias[c0] : 0.0f;
  const float bv1 = bias ? bias[c1] : 0.0f;
#pragma unroll
  for (int i = 0; i < 8; ++i) {
    const int rowA = rbase + i;
    const int rowB = rowA + 16;
    if (rowA < M) {
      float u = acc00[i] + bv0;
      float v = acc01[i] + bv1;
      if (act == 1) { u = fmaxf(u, 0.0f); v = fmaxf(v, 0.0f); }
      Cmat[(long)rowA * N + c0] = u;
      Cmat[(long)rowA * N + c1] = v;
    }
    if (rowB < M) {
      float u = acc10[i] + bv0;
      float v = acc11[i] + bv1;
      if (act == 1) { u = fmaxf(u, 0.0f); v = fmaxf(v, 0.0f); }
      Cmat[(long)rowB * N + c0] = u;
      Cmat[(long)rowB * N + c1] = v;
    }
  }
}

// ---------------------------------------------------------------------------
// Flash-style self-attention. One wave per (b, h, 16-query block).
// S tiles computed twice with WMMA (pass1 row-max, pass2 exp -> f16 P in LDS),
// then O = P @ V from LDS.  All loads unpredicated (clamped pointers); key
// masking done with VALU selects on the accumulator.
// ---------------------------------------------------------------------------
__global__ void __launch_bounds__(32) attn_kernel(
    const float* __restrict__ q, const float* __restrict__ k,
    const float* __restrict__ v, float* __restrict__ o)
{
  __shared__ _Float16 P[16 * 1024];   // 32 KB probability tile (f16)
  __shared__ float rowinv[16];

  const int lane = threadIdx.x;
  const int qt = blockIdx.x % 63;
  const int bh = blockIdx.x / 63;
  const int h = bh % NHh;
  const int b = bh / NHh;
  const int q0 = qt * 16;
  const int r  = lane & 15;
  const int kb = (lane >> 4) << 3;
  const float scale = 0.17677669529663687f;  // 1/sqrt(32)

  // A fragment: 16 query rows x head dim (K=32) -> one WMMA per S tile
  const int rq = (q0 + r < NQq) ? (q0 + r) : (NQq - 1);
  const v16h af = frag_f32(q + (long)(b * NQq + rq) * Cdim + h * HDd + kb);

  const float* kbase = k + ((long)b * NQq) * Cdim + h * HDd + kb;
  float rmax[8];
#pragma unroll
  for (int i = 0; i < 8; ++i) rmax[i] = -1e30f;

  // ---- pass 1: row maxima ----
  for (int kt = 0; kt < 63; ++kt) {
    const int c0 = kt * 16;
    const bool cok = (c0 + r) < NQq;
    const int ck = cok ? (c0 + r) : (NQq - 1);
    const v16h bf = frag_f32(kbase + (long)ck * Cdim);
    v8f s = {};
    s = __builtin_amdgcn_wmma_f32_16x16x32_f16(false, af, false, bf,
                                               (short)0, s, false, false);
#pragma unroll
    for (int i = 0; i < 8; ++i) {
      float m = cok ? s[i] * scale : -1e30f;
#pragma unroll
      for (int d = 1; d < 16; d <<= 1) m = fmaxf(m, __shfl_xor(m, d, 32));
      rmax[i] = fmaxf(rmax[i], m);
    }
  }

  // ---- pass 2: exp -> P (f16, LDS) ----
  for (int kt = 0; kt < 63; ++kt) {
    const int c0 = kt * 16;
    const bool cok = (c0 + r) < NQq;
    const int ck = cok ? (c0 + r) : (NQq - 1);
    const v16h bf = frag_f32(kbase + (long)ck * Cdim);
    v8f s = {};
    s = __builtin_amdgcn_wmma_f32_16x16x32_f16(false, af, false, bf,
                                               (short)0, s, false, false);
#pragma unroll
    for (int i = 0; i < 8; ++i) {
      const int row = ((lane >> 4) << 3) + i;
      const float e = cok ? __expf(s[i] * scale - rmax[i]) : 0.0f;
      P[row * 1024 + c0 + r] = (_Float16)e;
    }
  }
  // zero K-padding columns 1000..1023
  for (int idx = lane; idx < 16 * 24; idx += 32)
    P[(idx / 24) * 1024 + 1000 + (idx % 24)] = (_Float16)0.f;
  __syncthreads();

  if (lane < 16) {
    float sum = 0.f;
    const v8h* pr = (const v8h*)(P + lane * 1024);
    for (int j = 0; j < 128; ++j) {       // ds_load_b128 sweep (incl. zero pad)
      const v8h e = pr[j];
#pragma unroll
      for (int t = 0; t < 8; ++t) sum += (float)e[t];
    }
    rowinv[lane] = 1.0f / sum;
  }
  __syncthreads();

  // ---- O = P @ V : two N-half WMMAs per 32-key step ----
  const float* vbase = v + ((long)b * NQq) * Cdim + h * HDd;
  v8f o0 = {}, o1 = {};
  for (int ks = 0; ks < 1024; ks += 32) {
    const _Float16* pp = P + r * 1024 + ks + kb;
    const v8h plo = *(const v8h*)(pp);
    const v8h phi = *(const v8h*)(pp + 16);
    const v16h pf = __builtin_shufflevector(plo, phi, 0, 1, 2, 3, 4, 5, 6, 7,
                                            8, 9, 10, 11, 12, 13, 14, 15);
    v16h vf0, vf1;
#pragma unroll
    for (int i = 0; i < 8; ++i) {
      int k1 = ks + kb + i;       if (k1 >= NQq) k1 = NQq - 1;  // P cols are 0 there
      int k2 = ks + 16 + kb + i;  if (k2 >= NQq) k2 = NQq - 1;
      const float* v1 = vbase + (long)k1 * Cdim;
      const float* v2 = vbase + (long)k2 * Cdim;
      vf0[i]     = (_Float16)v1[r];
      vf0[8 + i] = (_Float16)v2[r];
      vf1[i]     = (_Float16)v1[16 + r];
      vf1[8 + i] = (_Float16)v2[16 + r];
    }
    o0 = __builtin_amdgcn_wmma_f32_16x16x32_f16(false, pf, false, vf0, (short)0, o0, false, false);
    o1 = __builtin_amdgcn_wmma_f32_16x16x32_f16(false, pf, false, vf1, (short)0, o1, false, false);
  }

#pragma unroll
  for (int i = 0; i < 8; ++i) {
    const int rr = ((lane >> 4) << 3) + i;
    const int row = q0 + rr;
    if (row < NQq) {
      const float inv = rowinv[rr];
      float* op = o + ((long)(b * NQq + row) * Cdim + h * HDd);
      op[r]      = o0[i] * inv;
      op[16 + r] = o1[i] * inv;
    }
  }
}

// ---------------------------------------------------------------------------
// Fused residual + LayerNorm, one wave per 256-wide row.
// ---------------------------------------------------------------------------
__global__ void __launch_bounds__(256) add_ln_kernel(
    const float* __restrict__ a, const float* __restrict__ res,
    const float* __restrict__ g, const float* __restrict__ beta,
    float* __restrict__ out, int M)
{
  const int lane = threadIdx.x & 31;
  const int wave = threadIdx.x >> 5;
  const long row = (long)blockIdx.x * 8 + wave;
  if (row >= M) return;
  const float* ap = a + row * Cdim;
  const float* rp = res ? res + row * Cdim : nullptr;
  float x[8];
  float s = 0.f;
#pragma unroll
  for (int i = 0; i < 8; ++i) {
    float v = ap[lane + i * 32];
    if (rp) v += rp[lane + i * 32];
    x[i] = v; s += v;
  }
#pragma unroll
  for (int d = 16; d >= 1; d >>= 1) s += __shfl_xor(s, d, 32);
  const float mean = s * (1.0f / 256.0f);
  float vs = 0.f;
#pragma unroll
  for (int i = 0; i < 8; ++i) { const float dv = x[i] - mean; vs += dv * dv; }
#pragma unroll
  for (int d = 16; d >= 1; d >>= 1) vs += __shfl_xor(vs, d, 32);
  const float rstd = rsqrtf(vs * (1.0f / 256.0f) + 1e-5f);
  float* op = out + row * Cdim;
#pragma unroll
  for (int i = 0; i < 8; ++i) {
    const int c = lane + i * 32;
    op[c] = (x[i] - mean) * rstd * g[c] + beta[c];
  }
}

// ---------------------------------------------------------------------------
// Deformable-attention bilinear sampling. One wave per (b, q, h); lane = d.
// ---------------------------------------------------------------------------
__global__ void __launch_bounds__(256) msda_sample_kernel(
    const float* __restrict__ vproj, const float* __restrict__ off,
    const float* __restrict__ awl, const float* __restrict__ ref,
    float* __restrict__ out)
{
  const int lane = threadIdx.x & 31;
  const int wave = threadIdx.x >> 5;
  const long gidx = (long)blockIdx.x * 8 + wave;
  if (gidx >= (long)Bb * NQq * NHh) return;
  const int h = (int)(gidx % NHh);
  const long bq = gidx / NHh;
  const int b = (int)(bq / NQq);

  const int HW[4][2] = {{100, 100}, {50, 50}, {25, 25}, {13, 13}};
  const int LSI[4] = {0, 10000, 12500, 13125};

  // softmax over the 16 sampling-weight logits (redundant per lane — cheap)
  float aw[16];
  const float* ap = awl + bq * (NHh * NLl * NPp) + h * (NLl * NPp);
  float m = -1e30f;
#pragma unroll
  for (int i = 0; i < 16; ++i) { aw[i] = ap[i]; m = fmaxf(m, aw[i]); }
  float s = 0.f;
#pragma unroll
  for (int i = 0; i < 16; ++i) { aw[i] = __expf(aw[i] - m); s += aw[i]; }
  const float inv = 1.0f / s;

  const float* op   = off + bq * Cdim + h * (NLl * NPp * 2);
  const float* refp = ref + bq * (NLl * 2);
  const float* vb   = vproj + ((long)b * NVv) * Cdim + h * HDd + lane;

  float acc = 0.f;
#pragma unroll
  for (int l = 0; l < NLl; ++l) {
    const int Hl = HW[l][0], Wl = HW[l][1];
    const float fW = (float)Wl, fH = (float)Hl;
    const float rx = refp[l * 2 + 0], ry = refp[l * 2 + 1];
    const float* vl = vb + (long)LSI[l] * Cdim;
#pragma unroll
    for (int p = 0; p < NPp; ++p) {
      const float ox = op[l * (NPp * 2) + p * 2 + 0];
      const float oy = op[l * (NPp * 2) + p * 2 + 1];
      const float x = (rx + ox / fW) * fW - 0.5f;
      const float y = (ry + oy / fH) * fH - 0.5f;
      const float x0f = floorf(x), y0f = floorf(y);
      const float lx = x - x0f, ly = y - y0f;
      const int x0 = (int)x0f, y0 = (int)y0f;
      float sv = 0.f;
#pragma unroll
      for (int cy = 0; cy < 2; ++cy) {
#pragma unroll
        for (int cx = 0; cx < 2; ++cx) {
          const int xi = x0 + cx, yi = y0 + cy;
          if (xi >= 0 && xi < Wl && yi >= 0 && yi < Hl) {
            const float w = (cx ? lx : 1.f - lx) * (cy ? ly : 1.f - ly);
            sv += w * vl[(long)(yi * Wl + xi) * Cdim];
          }
        }
      }
      acc += sv * aw[l * NPp + p] * inv;
    }
  }
  out[bq * Cdim + h * HDd + lane] = acc;
}

// ---------------------------------------------------------------------------
// Small elementwise kernels
// ---------------------------------------------------------------------------
__global__ void add_kernel(const float* __restrict__ a, const float* __restrict__ b,
                           float* __restrict__ o, long n) {
  const long i = (long)blockIdx.x * blockDim.x + threadIdx.x;
  if (i < n) o[i] = a[i] + b[i];
}

__global__ void silu_kernel(const float* __restrict__ a, float* __restrict__ o, int n) {
  const int i = blockIdx.x * blockDim.x + threadIdx.x;
  if (i < n) { const float x = a[i]; o[i] = x / (1.0f + __expf(-x)); }
}

__global__ void scaleshift_kernel(const float* __restrict__ ss, float* __restrict__ x) {
  const int i = blockIdx.x * blockDim.x + threadIdx.x;
  if (i >= Bb * NDNn * Cdim) return;
  const int c = i % Cdim;
  const int qn = (i / Cdim) % NDNn;
  const int b = i / (NDNn * Cdim);
  const float sc = ss[b * 2 * Cdim + c] + 1.0f;
  const float sh = ss[b * 2 * Cdim + Cdim + c];
  const long xi = ((long)(b * NQq + qn)) * Cdim + c;
  x[xi] = x[xi] * sc + sh;
}

// ---------------------------------------------------------------------------
// Orchestration
// ---------------------------------------------------------------------------
static inline void launch_gemm(const float* A, const float* W, const float* bias,
                               float* Cm, long M, int N, int K, int act,
                               hipStream_t stream) {
  const long tiles = ((M + 31) / 32) * (long)(N / 32);
  const int blocks = (int)((tiles + GEMM_WAVES - 1) / GEMM_WAVES);
  gemm_wmma_kernel<<<blocks, 256, 0, stream>>>(A, W, bias, Cm, (int)M, N, K, act);
}

extern "C" void kernel_launch(void* const* d_in, const int* in_sizes, int n_in,
                              void* d_out, int out_size, void* d_ws, size_t ws_size,
                              hipStream_t stream) {
  const float* query     = (const float*)d_in[0];
  const float* query_pos = (const float*)d_in[1];
  const float* refpts    = (const float*)d_in[2];
  const float* value     = (const float*)d_in[3];
  const float* time_dim  = (const float*)d_in[4];
  // d_in[5..7]: spatial_shapes / level_start_index / num_denoising (static, hardcoded)
  const float* sa_in_w  = (const float*)d_in[8];
  const float* sa_in_b  = (const float*)d_in[9];
  const float* sa_out_w = (const float*)d_in[10];
  const float* sa_out_b = (const float*)d_in[11];
  const float* n1_g = (const float*)d_in[12];
  const float* n1_b = (const float*)d_in[13];
  const float* n2_g = (const float*)d_in[14];
  const float* n2_b = (const float*)d_in[15];
  const float* n3_g = (const float*)d_in[16];
  const float* n3_b = (const float*)d_in[17];
  const float* val_w = (const float*)d_in[18];
  const float* val_b = (const float*)d_in[19];
  const float* off_w = (const float*)d_in[20];
  const float* off_b = (const float*)d_in[21];
  const float* aw_w  = (const float*)d_in[22];
  const float* aw_b  = (const float*)d_in[23];
  const float* co_w  = (const float*)d_in[24];
  const float* co_b  = (const float*)d_in[25];
  const float* f1_w  = (const float*)d_in[26];
  const float* f1_b  = (const float*)d_in[27];
  const float* f2_w  = (const float*)d_in[28];
  const float* f2_b  = (const float*)d_in[29];
  const float* t_w   = (const float*)d_in[30];
  const float* t_b   = (const float*)d_in[31];

  float* ws = (float*)d_ws;
  const long SZQ = (long)Bb * NQq * Cdim;     // 2,048,000
  float* qk    = ws;
  float* qpj   = ws + 1 * SZQ;
  float* kpj   = ws + 2 * SZQ;
  float* vpj   = ws + 3 * SZQ;
  float* sacat = ws + 4 * SZQ;
  float* sa    = ws + 5 * SZQ;
  float* x1    = ws + 6 * SZQ;
  float* q2    = ws + 7 * SZQ;
  float* offb  = ws + 8 * SZQ;
  float* samp  = ws + 9 * SZQ;
  float* ca    = ws + 10 * SZQ;
  float* x2    = ws + 11 * SZQ;
  float* yb    = ws + 12 * SZQ;
  float* awb   = ws + 13 * SZQ;                       // 1,024,000
  float* hb    = awb + (long)Bb * NQq * (NHh * NLl * NPp);
  float* vprojb = hb + (long)Bb * NQq * DFFd;         // 27,226,112
  float* stdb  = vprojb + (long)Bb * NVv * Cdim;      // 8,192
  float* ssb   = stdb + Bb * 4 * Cdim;                // 4,096

  const long M = (long)Bb * NQq;  // 8000
  const int EW = 256;

  // 1) qk = query + query_pos
  add_kernel<<<(int)((SZQ + EW - 1) / EW), EW, 0, stream>>>(query, query_pos, qk, SZQ);
  // 2) QKV projections (sa_in_w is [3C, C] row-major: q/k/v weight chunks)
  launch_gemm(qk,    sa_in_w,                   sa_in_b,            qpj, M, Cdim, Cdim, 0, stream);
  launch_gemm(qk,    sa_in_w + 1 * Cdim * Cdim, sa_in_b + Cdim,     kpj, M, Cdim, Cdim, 0, stream);
  launch_gemm(query, sa_in_w + 2 * Cdim * Cdim, sa_in_b + 2 * Cdim, vpj, M, Cdim, Cdim, 0, stream);
  // 3) flash self-attention
  attn_kernel<<<Bb * NHh * 63, 32, 0, stream>>>(qpj, kpj, vpj, sacat);
  // 4) output projection + residual LN (n2)
  launch_gemm(sacat, sa_out_w, sa_out_b, sa, M, Cdim, Cdim, 0, stream);
  add_ln_kernel<<<(int)((M + 7) / 8), EW, 0, stream>>>(sa, query, n2_g, n2_b, x1, (int)M);
  // 5) deformable cross-attention
  add_kernel<<<(int)((SZQ + EW - 1) / EW), EW, 0, stream>>>(x1, query_pos, q2, SZQ);
  launch_gemm(value, val_w, val_b, vprojb, (long)Bb * NVv, Cdim, Cdim, 0, stream);
  launch_gemm(q2, off_w, off_b, offb, M, NHh * NLl * NPp * 2, Cdim, 0, stream);
  launch_gemm(q2, aw_w,  aw_b,  awb,  M, NHh * NLl * NPp,     Cdim, 0, stream);
  {
    const long waves = (long)Bb * NQq * NHh;  // 64000
    msda_sample_kernel<<<(int)((waves + 7) / 8), EW, 0, stream>>>(vprojb, offb, awb, refpts, samp);
  }
  launch_gemm(samp, co_w, co_b, ca, M, Cdim, Cdim, 0, stream);
  add_ln_kernel<<<(int)((M + 7) / 8), EW, 0, stream>>>(ca, x1, n1_g, n1_b, x2, (int)M);
  // 6) time-conditioned scale/shift on denoising queries
  silu_kernel<<<(Bb * 4 * Cdim + EW - 1) / EW, EW, 0, stream>>>(time_dim, stdb, Bb * 4 * Cdim);
  launch_gemm(stdb, t_w, t_b, ssb, Bb, 2 * Cdim, 4 * Cdim, 0, stream);
  scaleshift_kernel<<<(Bb * NDNn * Cdim + EW - 1) / EW, EW, 0, stream>>>(ssb, x2);
  // 7) FFN + final residual LN (n3) -> d_out
  launch_gemm(x2, f1_w, f1_b, hb, M, DFFd, Cdim, 1, stream);
  launch_gemm(hb, f2_w, f2_b, yb, M, Cdim, DFFd, 0, stream);
  add_ln_kernel<<<(int)((M + 7) / 8), EW, 0, stream>>>(yb, x2, n3_g, n3_b, (float*)d_out, (int)M);
}
